// RNNG_61933428408490
// MI455X (gfx1250) — matrix operand order repaired and checked by reading, not compiled, not run on hardware
//
#include <hip/hip_runtime.h>

typedef __bf16 bf16_t;
typedef __attribute__((ext_vector_type(16))) __bf16 v16bf;
typedef __attribute__((ext_vector_type(8)))  __bf16 v8bf;
typedef __attribute__((ext_vector_type(8)))  float  v8f;

#define DIM   256      // embedding dim
#define HID   256      // LSTM hidden
#define MHID  256      // MLP hidden
#define NA    128      // actions
#define BATCH 64
#define TLEN  1024
#define G4    1024     // 4*HID
#define XDIM  768      // DIM + 2*HID

static __device__ __forceinline__ v16bf cat8(v8bf lo, v8bf hi) {
  return __builtin_shufflevector(lo, hi, 0,1,2,3,4,5,6,7,8,9,10,11,12,13,14,15);
}
static __device__ __forceinline__ v8f splat8(float v) {
  v8f r = {v,v,v,v,v,v,v,v}; return r;
}

// ---- CDNA5 async global->LDS copy (ASYNCcnt path), inline asm ------------
// LDS destination address comes from the dst VGPR; low 32 bits of a generic shared pointer are
// the LDS byte offset (addrspacecast local->flat keeps offset in low dword).
static __device__ __forceinline__ uint32_t lds_off(const void* p) {
  return (uint32_t)(uintptr_t)p;
}
static __device__ __forceinline__ void async_load_b128(uint32_t dst_lds, const void* src) {
  uint64_t ga = (uint64_t)(uintptr_t)src;
  asm volatile("global_load_async_to_lds_b128 %0, %1, off"
               :: "v"(dst_lds), "v"(ga) : "memory");
}
static __device__ __forceinline__ void wait_async0() {
  asm volatile("s_wait_asynccnt 0x0" ::: "memory");
}

// A fragment (16x32 bf16) from row-major LDS tile.
// ISA layout: lane = M; elems 0..7 = K = half*8+0..7, elems 8..15 = K = 16+half*8+0..7
static __device__ __forceinline__ v16bf load_a_tile(const bf16_t* base, int stride) {
  int lane = threadIdx.x & 31;
  int m    = lane & 15;
  int half = lane >> 4;
  const bf16_t* p = base + m * stride + half * 8;
  v8bf lo = *(const v8bf*)p;          // 16B LDS load
  v8bf hi = *(const v8bf*)(p + 16);   // 16B LDS load
  return cat8(lo, hi);
}

// B fragment from pre-swizzled weights: layout [kt][nt][lane][16 bf16] contiguous (32B/lane)
static __device__ __forceinline__ v16bf load_b_tile(const bf16_t* swz, int kt, int nt, int NT) {
  int lane = threadIdx.x & 31;
  const bf16_t* p = swz + ((((size_t)kt * NT) + nt) * 32 + (size_t)lane) * 16;
  v8bf lo = *(const v8bf*)p;
  v8bf hi = *(const v8bf*)(p + 8);
  return cat8(lo, hi);
}

static __device__ __forceinline__ v8f wmma_bf16(v16bf a, v16bf b, v8f c) {
  return __builtin_amdgcn_wmma_f32_16x16x32_bf16(false, a, false, b, (short)0, c, false, false);
}
static __device__ __forceinline__ float sigmoidf_(float x) {
  return 1.0f / (1.0f + __expf(-x));
}
static __device__ __forceinline__ v8bf cvt8(float4 a, float4 b) {
  v8bf r = {(bf16_t)a.x,(bf16_t)a.y,(bf16_t)a.z,(bf16_t)a.w,
            (bf16_t)b.x,(bf16_t)b.y,(bf16_t)b.z,(bf16_t)b.w};
  return r;
}

// ---------------------------------------------------------------------------
// Pre-swizzle fp32 [K][N] row-major weights into WMMA B-fragment order (bf16):
// out[((kt*NT+nt)*32+lane)*16+e] = W[kt*32 + (e>>3)*16 + (lane>>4)*8 + (e&7)][nt*16 + (lane&15)]
// ---------------------------------------------------------------------------
__global__ __launch_bounds__(256)
void swizzle_w(const float* __restrict__ W, bf16_t* __restrict__ out, int K, int N) {
  int idx = blockIdx.x * 256 + threadIdx.x;
  if (idx >= K * N) return;
  int e    = idx & 15;
  int lane = (idx >> 4) & 31;
  int NT   = N >> 4;
  int nt   = (idx >> 9) % NT;
  int kt   = idx / (NT << 9);
  int k = kt * 32 + (e >> 3) * 16 + (lane >> 4) * 8 + (e & 7);
  int n = nt * 16 + (lane & 15);
  out[idx] = (bf16_t)W[(size_t)k * N + n];
}

// ---------------------------------------------------------------------------
// Persistent LSTM scan. Grid = 8 WGs: blockIdx = lstm*4 + batch_tile.
// 256 threads = 8 waves; wave w owns hidden cols [w*32,w*32+32); cell state in
// VGPRs for the whole scan; h exchanged via LDS. Embedding gather for step t+1
// is issued as GLOBAL_LOAD_ASYNC_TO_LDS_B128 and overlaps step t's WMMAs.
// Stores h(t) into hs[..][t+1][..] (time-shifted), hs[..][0][..] = 0.
// ---------------------------------------------------------------------------
__global__ __launch_bounds__(256)
void lstm_kernel(const int* __restrict__ stack_ids, const int* __restrict__ action_ids,
                 const float* __restrict__ word_emb, const float* __restrict__ action_emb,
                 const bf16_t* __restrict__ Wsx, const bf16_t* __restrict__ Wsh,
                 const float* __restrict__ bs,
                 const bf16_t* __restrict__ Whx, const bf16_t* __restrict__ Whh,
                 const float* __restrict__ bh,
                 bf16_t* __restrict__ hs_s, bf16_t* __restrict__ hs_h)
{
  const int lstm  = blockIdx.x >> 2;
  const int btile = blockIdx.x & 3;
  const int*    ids  = lstm ? action_ids : stack_ids;
  const float*  emb  = lstm ? action_emb : word_emb;
  const bf16_t* Wx   = lstm ? Whx : Wsx;
  const bf16_t* Wh   = lstm ? Whh : Wsh;
  const float*  bias = lstm ? bh  : bs;
  bf16_t*       hs   = lstm ? hs_h : hs_s;

  __shared__ __align__(16) float  x_stage[2][16 * DIM];  // 32KB async landing pad
  __shared__ __align__(16) bf16_t x_tile[16 * DIM];      // 8KB
  __shared__ __align__(16) bf16_t h_tile[16 * HID];      // 8KB

  const int tid  = threadIdx.x;
  const int wave = tid >> 5;
  const int lane = tid & 31;
  const int nlo  = lane & 15;
  const int half = lane >> 4;

  for (int i = tid; i < 16 * HID; i += 256) h_tile[i] = (bf16_t)0.0f;
  for (int i = tid; i < 16 * HID; i += 256) {           // hs[.., t=0, ..] = 0
    int r = i >> 8, c = i & (HID - 1);
    hs[(size_t)(btile * 16 + r) * TLEN * HID + c] = (bf16_t)0.0f;
  }

  v8f c0 = splat8(0.0f), c1 = splat8(0.0f);             // cell state, 16 rows x 32 cols

  float bfrag[8];
  #pragma unroll
  for (int f = 0; f < 8; ++f) {                         // f = gate*2 + sub
    int g = f >> 1, sub = f & 1;
    bfrag[f] = bias[g * HID + wave * 32 + sub * 16 + nlo];
  }

  const int gr = tid >> 4;            // gather row 0..15
  const int gc = (tid & 15) << 4;     // gather col base (16 floats)
  const uint32_t stage0 = lds_off(&x_stage[0][0]);

  auto issue_gather = [&](int t, int buf) {
    int id = ids[(size_t)(btile * 16 + gr) * TLEN + t];
    const float* src = emb + (size_t)id * DIM + gc;
    uint32_t dst = stage0 + (uint32_t)((buf * 16 * DIM + gr * DIM + gc) * 4);
    #pragma unroll
    for (int j = 0; j < 16; j += 4)
      async_load_b128(dst + (uint32_t)(j * 4), src + j);   // 4 x 16B per thread
  };

  issue_gather(0, 0);

  for (int t = 0; t < TLEN; ++t) {
    wait_async0();       // my wave's async gather for step t has landed
    __syncthreads();     // all waves' gathers landed; prev h_tile writes visible

    { // convert staged fp32 -> bf16 x_tile (LDS->LDS, cheap)
      const float* src = &x_stage[t & 1][gr * DIM + gc];
      bf16_t*      dst = x_tile + gr * DIM + gc;
      float4 a0 = ((const float4*)src)[0];
      float4 a1 = ((const float4*)src)[1];
      float4 a2 = ((const float4*)src)[2];
      float4 a3 = ((const float4*)src)[3];
      ((v8bf*)dst)[0] = cvt8(a0, a1);
      ((v8bf*)dst)[1] = cvt8(a2, a3);
    }
    if (t + 1 < TLEN) issue_gather(t + 1, (t + 1) & 1);  // overlaps the WMMAs below
    __syncthreads();     // x_tile ready

    v8f acc[8];
    #pragma unroll
    for (int f = 0; f < 8; ++f) acc[f] = splat8(bfrag[f]);

    for (int kt = 0; kt < 8; ++kt) {
      v16bf ax = load_a_tile(x_tile + kt * 32, DIM);
      #pragma unroll
      for (int f = 0; f < 8; ++f) {
        int g = f >> 1, sub = f & 1;
        int nt = (g * HID + wave * 32 + sub * 16) >> 4;
        acc[f] = wmma_bf16(ax, load_b_tile(Wx, kt, nt, G4 / 16), acc[f]);
      }
      v16bf ah = load_a_tile(h_tile + kt * 32, HID);
      #pragma unroll
      for (int f = 0; f < 8; ++f) {
        int g = f >> 1, sub = f & 1;
        int nt = (g * HID + wave * 32 + sub * 16) >> 4;
        acc[f] = wmma_bf16(ah, load_b_tile(Wh, kt, nt, G4 / 16), acc[f]);
      }
    }
    __syncthreads();   // all reads of h_tile done before overwrite

    #pragma unroll
    for (int sub = 0; sub < 2; ++sub) {
      v8f gi = acc[0 + sub], gf = acc[2 + sub], gg = acc[4 + sub], go = acc[6 + sub];
      v8f& cc = sub ? c1 : c0;
      int col = wave * 32 + sub * 16 + nlo;
      #pragma unroll
      for (int r = 0; r < 8; ++r) {
        float cn = sigmoidf_(gf[r]) * cc[r] + sigmoidf_(gi[r]) * tanhf(gg[r]);
        cc[r] = cn;
        float hv = sigmoidf_(go[r]) * tanhf(cn);
        int row = r + 8 * half;
        bf16_t hb = (bf16_t)hv;
        h_tile[row * HID + col] = hb;
        if (t + 1 < TLEN)
          hs[((size_t)(btile * 16 + row) * TLEN + (t + 1)) * HID + col] = hb;
      }
    }
  }
}

// ---------------------------------------------------------------------------
// MLP head + log-softmax + NLL. Grid = 2048 WGs x 64 threads (2 waves),
// each WG owns 32 flattened (b,t) rows. h_t/s_t slabs arrive via async LDS
// copies overlapped with the buffer-embedding gather. logits overlay xcat.
// ---------------------------------------------------------------------------
__global__ __launch_bounds__(64)
void mlp_kernel(const int* __restrict__ buffer_ids, const int* __restrict__ action_ids,
                const float* __restrict__ word_emb,
                const bf16_t* __restrict__ hs_s, const bf16_t* __restrict__ hs_h,
                const bf16_t* __restrict__ W1s, const float* __restrict__ b1,
                const bf16_t* __restrict__ W2s, const float* __restrict__ b2,
                float* __restrict__ partials)
{
  __shared__ __align__(16) bf16_t xcat[32 * XDIM];   // 48KB
  __shared__ __align__(16) bf16_t ybuf[32 * MHID];   // 16KB
  float* lg     = (float*)xcat;                      // overlay: 32*128 f32 = 16KB
  float* rownll = (float*)xcat + 32 * NA;            // next 128B, still inside xcat

  const int tid  = threadIdx.x;
  const int wave = tid >> 5;
  const int lane = tid & 31;
  const int nlo  = lane & 15;
  const int half = lane >> 4;
  const size_t base = (size_t)blockIdx.x * 32;       // flat row = b*T + t
  const uint32_t xcat0 = lds_off(xcat);

  // async copy h_t / s_t slabs (bf16, 16B-aligned) straight into xcat
  for (int i = tid * 8; i < 32 * HID; i += 64 * 8) {
    int r = i >> 8, c = i & (HID - 1);
    async_load_b128(xcat0 + (uint32_t)((r * XDIM + DIM + c) * 2),
                    hs_h + (base + r) * HID + c);
    async_load_b128(xcat0 + (uint32_t)((r * XDIM + DIM + HID + c) * 2),
                    hs_s + (base + r) * HID + c);
  }
  // overlapped: gather buffer-top embeddings (needs f32->bf16 convert)
  for (int i = tid * 4; i < 32 * DIM; i += 64 * 4) {
    int r = i >> 8, c = i & (DIM - 1);
    int id = buffer_ids[base + r];
    float4 v = *(const float4*)(word_emb + (size_t)id * DIM + c);
    bf16_t* d = xcat + r * XDIM + c;
    d[0] = (bf16_t)v.x; d[1] = (bf16_t)v.y; d[2] = (bf16_t)v.z; d[3] = (bf16_t)v.w;
  }
  wait_async0();
  __syncthreads();

  // layer 1: relu(xcat @ W1 + b1) -> ybuf (bf16)
  const bf16_t* arow = xcat + (wave * 16) * XDIM;
  for (int ntg = 0; ntg < 4; ++ntg) {
    v8f acc[4];
    #pragma unroll
    for (int j = 0; j < 4; ++j) acc[j] = splat8(b1[(ntg * 4 + j) * 16 + nlo]);
    for (int kt = 0; kt < 24; ++kt) {
      v16bf a = load_a_tile(arow + kt * 32, XDIM);
      #pragma unroll
      for (int j = 0; j < 4; ++j)
        acc[j] = wmma_bf16(a, load_b_tile(W1s, kt, ntg * 4 + j, MHID / 16), acc[j]);
    }
    #pragma unroll
    for (int j = 0; j < 4; ++j) {
      int nb = (ntg * 4 + j) * 16 + nlo;
      #pragma unroll
      for (int r = 0; r < 8; ++r) {
        int row = wave * 16 + r + 8 * half;
        ybuf[row * MHID + nb] = (bf16_t)fmaxf(acc[j][r], 0.0f);
      }
    }
  }
  __syncthreads();   // everyone done with xcat before lg overlays it

  // layer 2: logits = ybuf @ W2 + b2 -> lg (f32, overlay)
  const bf16_t* yrow = ybuf + (wave * 16) * MHID;
  for (int ntg = 0; ntg < 2; ++ntg) {
    v8f acc[4];
    #pragma unroll
    for (int j = 0; j < 4; ++j) acc[j] = splat8(b2[(ntg * 4 + j) * 16 + nlo]);
    for (int kt = 0; kt < 8; ++kt) {
      v16bf a = load_a_tile(yrow + kt * 32, MHID);
      #pragma unroll
      for (int j = 0; j < 4; ++j)
        acc[j] = wmma_bf16(a, load_b_tile(W2s, kt, ntg * 4 + j, NA / 16), acc[j]);
    }
    #pragma unroll
    for (int j = 0; j < 4; ++j) {
      int nb = (ntg * 4 + j) * 16 + nlo;
      #pragma unroll
      for (int r = 0; r < 8; ++r) {
        int row = wave * 16 + r + 8 * half;
        lg[row * NA + nb] = acc[j][r];
      }
    }
  }
  __syncthreads();

  // per-row NLL: logsumexp(logits) - logits[action]
  if (tid < 32) {
    const float* l = lg + tid * NA;
    float mx = l[0];
    for (int j = 1; j < NA; ++j) mx = fmaxf(mx, l[j]);
    float s = 0.0f;
    for (int j = 0; j < NA; ++j) s += __expf(l[j] - mx);
    int aid = action_ids[base + tid];
    rownll[tid] = __logf(s) + mx - l[aid];
  }
  __syncthreads();
  if (tid == 0) {   // deterministic in-order block partial
    float s = 0.0f;
    for (int r = 0; r < 32; ++r) s += rownll[r];
    partials[blockIdx.x] = s;
  }
}

__global__ void reduce_kernel(const float* __restrict__ partials, int n, float* __restrict__ out) {
  if (threadIdx.x == 0 && blockIdx.x == 0) {
    float s = 0.0f;
    for (int i = 0; i < n; ++i) s += partials[i];   // fixed order -> deterministic
    out[0] = s;
  }
}

// ---------------------------------------------------------------------------
extern "C" void kernel_launch(void* const* d_in, const int* in_sizes, int n_in,
                              void* d_out, int out_size, void* d_ws, size_t ws_size,
                              hipStream_t stream)
{
  const int*   stack_ids  = (const int*)  d_in[0];
  const int*   buffer_ids = (const int*)  d_in[1];
  const int*   action_ids = (const int*)  d_in[2];
  const float* word_emb   = (const float*)d_in[3];
  const float* action_emb = (const float*)d_in[4];
  const float* Wsx = (const float*)d_in[5];
  const float* Wsh = (const float*)d_in[6];
  const float* bs  = (const float*)d_in[7];
  const float* Whx = (const float*)d_in[8];
  const float* Whh = (const float*)d_in[9];
  const float* bh  = (const float*)d_in[10];
  const float* W1  = (const float*)d_in[11];
  const float* b1  = (const float*)d_in[12];
  const float* W2  = (const float*)d_in[13];
  const float* b2  = (const float*)d_in[14];

  char* ws = (char*)d_ws;
  size_t off = 0;
  auto carve = [&](size_t bytes) -> void* {
    void* p = ws + off;
    off += (bytes + 255) & ~(size_t)255;
    return p;
  };
  bf16_t* Wsx_s = (bf16_t*)carve((size_t)DIM  * G4   * 2);
  bf16_t* Wsh_s = (bf16_t*)carve((size_t)HID  * G4   * 2);
  bf16_t* Whx_s = (bf16_t*)carve((size_t)DIM  * G4   * 2);
  bf16_t* Whh_s = (bf16_t*)carve((size_t)HID  * G4   * 2);
  bf16_t* W1_s  = (bf16_t*)carve((size_t)XDIM * MHID * 2);
  bf16_t* W2_s  = (bf16_t*)carve((size_t)MHID * NA   * 2);
  bf16_t* hs_s  = (bf16_t*)carve((size_t)BATCH * TLEN * HID * 2);   // ~33.5 MB
  bf16_t* hs_h  = (bf16_t*)carve((size_t)BATCH * TLEN * HID * 2);   // ~33.5 MB
  const int nblk = (BATCH * TLEN) / 32;                             // 2048
  float*  partials = (float*)carve((size_t)nblk * 4);
  (void)ws_size; (void)in_sizes; (void)n_in; (void)out_size;

  swizzle_w<<<(DIM  * G4  ) / 256, 256, 0, stream>>>(Wsx, Wsx_s, DIM,  G4);
  swizzle_w<<<(HID  * G4  ) / 256, 256, 0, stream>>>(Wsh, Wsh_s, HID,  G4);
  swizzle_w<<<(DIM  * G4  ) / 256, 256, 0, stream>>>(Whx, Whx_s, DIM,  G4);
  swizzle_w<<<(HID  * G4  ) / 256, 256, 0, stream>>>(Whh, Whh_s, HID,  G4);
  swizzle_w<<<(XDIM * MHID) / 256, 256, 0, stream>>>(W1,  W1_s,  XDIM, MHID);
  swizzle_w<<<(MHID * NA  ) / 256, 256, 0, stream>>>(W2,  W2_s,  MHID, NA);

  lstm_kernel<<<8, 256, 0, stream>>>(stack_ids, action_ids, word_emb, action_emb,
                                     Wsx_s, Wsh_s, bs, Whx_s, Whh_s, bh, hs_s, hs_h);

  mlp_kernel<<<nblk, 64, 0, stream>>>(buffer_ids, action_ids, word_emb,
                                      hs_s, hs_h, W1_s, b1, W2_s, b2, partials);

  reduce_kernel<<<1, 32, 0, stream>>>(partials, nblk, (float*)d_out);
}
